// CausalMultiheadAttention_84713934946713
// MI455X (gfx1250) — matrix operand order repaired and checked by reading, not compiled, not run on hardware
//
#include <hip/hip_runtime.h>
#include <hip/hip_bf16.h>

typedef __bf16 bf16;
typedef __attribute__((ext_vector_type(16))) __bf16 v16bf;
typedef __attribute__((ext_vector_type(8)))  __bf16 v8bf;
typedef __attribute__((ext_vector_type(8)))  float  v8f;

#define B_   4
#define T_   2048
#define C_   1024
#define NH_  16
#define DH_  64
#define LOG2E 1.44269504088896340736f

union AFrag { v16bf v; v8bf h[2]; };

__device__ __forceinline__ v16bf ld16g(const bf16* p) { return *reinterpret_cast<const v16bf*>(p); }
__device__ __forceinline__ v8bf  ld8g (const bf16* p) { return *reinterpret_cast<const v8bf*>(p); }

__device__ __forceinline__ v8f wmma_bf16(const v16bf& a, const v16bf& b, const v8f& c) {
  return __builtin_amdgcn_wmma_f32_16x16x32_bf16(false, a, false, b, (short)0, c, false, false);
}

// ---------------------------------------------------------------- convert
__global__ void cvt_f32_to_bf16(const float* __restrict__ in, bf16* __restrict__ out, int n) {
  int i = blockIdx.x * blockDim.x + threadIdx.x;
  int stride = gridDim.x * blockDim.x;
  for (; i < n; i += stride) out[i] = (bf16)in[i];
}

// ---------------------------------------------------------------- projection GEMM
// Y[M=8192, N=1024] = A @ W^T + bias. Per-wave 32x64 tile (8 WMMAs / K-step).
// 2x-unrolled K loop with two alternating register buffer sets: loads of one
// stage overlap WMMAs of the other, with NO register rotation copies.
// mode 0: bf16 out, heads layout (B,NH,T,DH)         (Q, K)
// mode 1: bf16 out, transposed heads (B,NH,DH,T)     (V)
// mode 2: f32 out, flat (M, C_)                      (final out-proj)
__launch_bounds__(128)
__global__ void proj_gemm(const bf16* __restrict__ A, const bf16* __restrict__ W,
                          const float* __restrict__ bias,
                          bf16* __restrict__ out_bf, float* __restrict__ out_f,
                          int mode, float scale) {
  const int lane = threadIdx.x & 31;
  const int wid  = threadIdx.x >> 5;
  const int task = blockIdx.x * (blockDim.x >> 5) + wid;
  const int NTL  = C_ / 64;                 // 16 column-tile groups
  const int m_base = (task / NTL) * 32;
  const int o_base = (task % NTL) * 64;
  const int hlf  = lane >> 4;               // 0|1 half-wave
  const int l16  = lane & 15;
  const int e0   = hlf * 16;                // A-frag K base within 0..31
  const int koff = hlf * 16;                // B-frag K base

  v8f acc[2][4];
  #pragma unroll
  for (int mi = 0; mi < 2; ++mi)
    #pragma unroll
    for (int i = 0; i < 4; ++i) acc[mi][i] = (v8f){0,0,0,0,0,0,0,0};

  const bf16* arow0 = A + (size_t)(m_base + l16) * C_;
  const bf16* arow1 = A + (size_t)(m_base + 16 + l16) * C_;
  const bf16* wr0 = W + (size_t)(o_base +  0 + l16) * C_;
  const bf16* wr1 = W + (size_t)(o_base + 16 + l16) * C_;
  const bf16* wr2 = W + (size_t)(o_base + 32 + l16) * C_;
  const bf16* wr3 = W + (size_t)(o_base + 48 + l16) * C_;

  AFrag aA[2], aB[2];          // stage A / stage B A-fragments
  v16bf bA[4], bB[4];          // stage A / stage B B-fragments

#define LOAD_STAGE(AF, BF, c)                                            \
  do {                                                                   \
    (AF)[0].h[0] = ld8g(arow0 + (c) + e0);                               \
    (AF)[0].h[1] = ld8g(arow0 + (c) + e0 + 8);                           \
    (AF)[1].h[0] = ld8g(arow1 + (c) + e0);                               \
    (AF)[1].h[1] = ld8g(arow1 + (c) + e0 + 8);                           \
    (BF)[0] = ld16g(wr0 + (c) + koff);                                   \
    (BF)[1] = ld16g(wr1 + (c) + koff);                                   \
    (BF)[2] = ld16g(wr2 + (c) + koff);                                   \
    (BF)[3] = ld16g(wr3 + (c) + koff);                                   \
  } while (0)

#define WMMA_STAGE(AF, BF)                                               \
  do {                                                                   \
    _Pragma("unroll")                                                    \
    for (int i = 0; i < 4; ++i) {                                        \
      acc[0][i] = wmma_bf16((AF)[0].v, (BF)[i], acc[0][i]);              \
      acc[1][i] = wmma_bf16((AF)[1].v, (BF)[i], acc[1][i]);              \
    }                                                                    \
  } while (0)

  LOAD_STAGE(aA, bA, 0);
  for (int c0 = 0; c0 < C_; c0 += 64) {
    __builtin_prefetch(arow0 + c0 + 512, 0, 1);        // global_prefetch_b8
    LOAD_STAGE(aB, bB, c0 + 32);                       // always in range (<= C_-32)
    WMMA_STAGE(aA, bA);
    if (c0 + 64 < C_) LOAD_STAGE(aA, bA, c0 + 64);
    WMMA_STAGE(aB, bB);
  }
#undef LOAD_STAGE
#undef WMMA_STAGE

  #pragma unroll
  for (int mi = 0; mi < 2; ++mi) {
    #pragma unroll
    for (int i = 0; i < 4; ++i) {
      const int col = o_base + i * 16 + l16;
      const float bv = bias[col];
      #pragma unroll
      for (int r = 0; r < 8; ++r) {
        const int m = m_base + mi * 16 + r + hlf * 8;   // C-layout row
        float val = (acc[mi][i][r] + bv) * scale;
        if (mode == 2) {
          out_f[(size_t)m * C_ + col] = val;
        } else {
          const int bidx = m >> 11;                     // m / T_
          const int t    = m & (T_ - 1);
          const int hd   = col >> 6;                    // head
          const int d    = col & 63;
          size_t idx;
          if (mode == 0) idx = (((size_t)(bidx * NH_ + hd) * T_) + t) * DH_ + d;
          else           idx = (((size_t)(bidx * NH_ + hd) * DH_) + d) * T_ + t;
          out_bf[idx] = (bf16)val;
        }
      }
    }
  }
}

// ---------------------------------------------------------------- causal flash attention
// One wave per (b, head, 16-row q tile). Q pre-scaled by 1/sqrt(D).
__launch_bounds__(128)
__global__ void attn_fwd(const bf16* __restrict__ Q, const bf16* __restrict__ K,
                         const bf16* __restrict__ Vt, bf16* __restrict__ ctx) {
  __shared__ __attribute__((aligned(64))) bf16 pbuf[4][16 * 32];   // per-wave 16x32 P tile

  const int lane = threadIdx.x & 31;
  const int wid  = threadIdx.x >> 5;
  const int task = blockIdx.x * (blockDim.x >> 5) + wid;
  const int QT   = T_ / 16;                  // 128 q tiles per (b,head)
  const int bn   = task / QT;
  const int t_base = (task % QT) * 16;
  const int b    = bn / NH_;
  const int hd   = bn % NH_;
  const int hlf  = lane >> 4;
  const int l16  = lane & 15;
  const int e0   = hlf * 16;
  const int koff = hlf * 16;

  const bf16* Qb = Q  + ((size_t)(b * NH_ + hd) * T_) * DH_;
  const bf16* Kb = K  + ((size_t)(b * NH_ + hd) * T_) * DH_;
  const bf16* Vb = Vt + ((size_t)(b * NH_ + hd) * DH_) * T_;

  // Q A-fragments for K-dim d = 0..31 and 32..63
  AFrag aq0, aq1;
  {
    const bf16* qr = Qb + (size_t)(t_base + l16) * DH_;
    aq0.h[0] = ld8g(qr + e0);        aq0.h[1] = ld8g(qr + e0 + 8);
    aq1.h[0] = ld8g(qr + 32 + e0);   aq1.h[1] = ld8g(qr + 32 + e0 + 8);
  }

  float mrow[8], lrow[8], alpha[8];
  v8f cacc[4];
  #pragma unroll
  for (int r = 0; r < 8; ++r) { mrow[r] = -1e30f; lrow[r] = 0.0f; }
  #pragma unroll
  for (int i = 0; i < 4; ++i) cacc[i] = (v8f){0,0,0,0,0,0,0,0};

  const int t_lane = t_base + hlf * 8;       // row of accumulator element r is t_lane + r

  for (int s0 = 0; s0 < t_base + 16; s0 += 32) {
    // ---- scores: two 16x16 tiles over s, K-dim = d = 64 (2 WMMAs each)
    v8f sc0 = (v8f){0,0,0,0,0,0,0,0};
    v8f sc1 = (v8f){0,0,0,0,0,0,0,0};
    {
      const bf16* kr0 = Kb + (size_t)(s0 + l16) * DH_;
      const bf16* kr1 = Kb + (size_t)(s0 + 16 + l16) * DH_;
      __builtin_prefetch(kr0 + 32 * DH_, 0, 1);        // next s-block K rows
      __builtin_prefetch(kr1 + 32 * DH_, 0, 1);
      sc0 = wmma_bf16(aq0.v, ld16g(kr0 + koff),      sc0);
      sc0 = wmma_bf16(aq1.v, ld16g(kr0 + 32 + koff), sc0);
      sc1 = wmma_bf16(aq0.v, ld16g(kr1 + koff),      sc1);
      sc1 = wmma_bf16(aq1.v, ld16g(kr1 + 32 + koff), sc1);
    }

    // ---- causal mask + online softmax (row stats via half-wave shfl reductions)
    const int col0 = s0 + l16;
    const int col1 = col0 + 16;
    #pragma unroll
    for (int r = 0; r < 8; ++r) {
      const int t = t_lane + r;
      float x0 = (col0 <= t) ? sc0[r] : -1e30f;
      float x1 = (col1 <= t) ? sc1[r] : -1e30f;
      float mx = fmaxf(x0, x1);
      mx = fmaxf(mx, __shfl_xor(mx, 1));
      mx = fmaxf(mx, __shfl_xor(mx, 2));
      mx = fmaxf(mx, __shfl_xor(mx, 4));
      mx = fmaxf(mx, __shfl_xor(mx, 8));
      const float mnew = fmaxf(mrow[r], mx);
      alpha[r] = __builtin_amdgcn_exp2f((mrow[r] - mnew) * LOG2E);
      const float p0 = __builtin_amdgcn_exp2f((x0 - mnew) * LOG2E);
      const float p1 = __builtin_amdgcn_exp2f((x1 - mnew) * LOG2E);
      float sum = p0 + p1;
      sum += __shfl_xor(sum, 1);
      sum += __shfl_xor(sum, 2);
      sum += __shfl_xor(sum, 4);
      sum += __shfl_xor(sum, 8);
      lrow[r] = lrow[r] * alpha[r] + sum;
      mrow[r] = mnew;
      const int prow = r + hlf * 8;          // C-layout row -> LDS row
      pbuf[wid][prow * 32 + l16]      = (bf16)p0;
      pbuf[wid][prow * 32 + 16 + l16] = (bf16)p1;
    }

    // rescale running context by alpha (per-row)
    #pragma unroll
    for (int i = 0; i < 4; ++i)
      #pragma unroll
      for (int r = 0; r < 8; ++r) cacc[i][r] *= alpha[r];

    // ---- P: C-layout -> A-layout via per-wave LDS tile (same-wave LDS is in-order)
    AFrag ap;
    ap.h[0] = *reinterpret_cast<const v8bf*>(&pbuf[wid][l16 * 32 + e0]);
    ap.h[1] = *reinterpret_cast<const v8bf*>(&pbuf[wid][l16 * 32 + e0 + 8]);

    // ---- ctx += P @ V  (V stored transposed: contiguous along s per column d)
    #pragma unroll
    for (int i = 0; i < 4; ++i) {
      const bf16* vr = Vb + (size_t)(i * 16 + l16) * T_ + s0 + koff;
      __builtin_prefetch(vr + 32, 0, 1);               // next s-block V columns
      cacc[i] = wmma_bf16(ap.v, ld16g(vr), cacc[i]);
    }
  }

  // ---- normalize, merge heads into (B, T, C) bf16
  #pragma unroll
  for (int i = 0; i < 4; ++i) {
    const int cdim = hd * DH_ + i * 16 + l16;
    #pragma unroll
    for (int r = 0; r < 8; ++r) {
      const int t = t_lane + r;
      const float val = cacc[i][r] / lrow[r];
      ctx[((size_t)b * T_ + t) * C_ + cdim] = (bf16)val;
    }
  }
}

// ---------------------------------------------------------------- launch
extern "C" void kernel_launch(void* const* d_in, const int* in_sizes, int n_in,
                              void* d_out, int out_size, void* d_ws, size_t ws_size,
                              hipStream_t stream) {
  const float* x  = (const float*)d_in[0];
  const float* Wq = (const float*)d_in[1];
  const float* bq = (const float*)d_in[2];
  const float* Wk = (const float*)d_in[3];
  const float* bk = (const float*)d_in[4];
  const float* Wv = (const float*)d_in[5];
  const float* bv = (const float*)d_in[6];
  const float* Wo = (const float*)d_in[7];
  const float* bo = (const float*)d_in[8];
  float* out = (float*)d_out;

  const size_t M1 = 1ull << 20;
  bf16* wsb  = (bf16*)d_ws;
  bf16* xb   = wsb;             // 8M elems  : x in bf16
  bf16* wqb  = wsb + 8 * M1;    // 1M each   : weights in bf16
  bf16* wkb  = wsb + 9 * M1;
  bf16* wvb  = wsb + 10 * M1;
  bf16* wob  = wsb + 11 * M1;
  bf16* qb   = wsb + 12 * M1;   // 8M : Q  (B,NH,T,DH), pre-scaled by 1/8
  bf16* kbuf = wsb + 20 * M1;   // 8M : K  (B,NH,T,DH)
  bf16* vtb  = wsb + 28 * M1;   // 8M : V^T (B,NH,DH,T)
  bf16* ctxb = wsb + 36 * M1;   // 8M : ctx (B,T,C)

  const int NX = B_ * T_ * C_;  // 8388608
  const int NW = C_ * C_;       // 1048576
  cvt_f32_to_bf16<<<4096, 256, 0, stream>>>(x,  xb,  NX);
  cvt_f32_to_bf16<<<1024, 256, 0, stream>>>(Wq, wqb, NW);
  cvt_f32_to_bf16<<<1024, 256, 0, stream>>>(Wk, wkb, NW);
  cvt_f32_to_bf16<<<1024, 256, 0, stream>>>(Wv, wvb, NW);
  cvt_f32_to_bf16<<<1024, 256, 0, stream>>>(Wo, wob, NW);

  // 8192/32 M-tiles * 1024/64 N-tiles = 4096 wave tasks; 4 waves/block
  const int gemm_blocks = 1024;
  const float qscale = 0.125f;  // 1/sqrt(64)
  proj_gemm<<<gemm_blocks, 128, 0, stream>>>(xb, wqb, bq, qb,   nullptr, 0, qscale);
  proj_gemm<<<gemm_blocks, 128, 0, stream>>>(xb, wkb, bk, kbuf, nullptr, 0, 1.0f);
  proj_gemm<<<gemm_blocks, 128, 0, stream>>>(xb, wvb, bv, vtb,  nullptr, 1, 1.0f);

  // 4*16*128 = 8192 wave tasks; 4 waves/block
  attn_fwd<<<2048, 128, 0, stream>>>(qb, kbuf, vtb, ctxb);

  proj_gemm<<<gemm_blocks, 128, 0, stream>>>(ctxb, wob, bo, nullptr, out, 2, 1.0f);
}
